// LinearQwen3VLMoeTextSparseMoeBlock_63763084476760
// MI455X (gfx1250) — matrix operand order repaired and checked by reading, not compiled, not run on hardware
//
#include <hip/hip_runtime.h>
#include <hip/hip_bf16.h>

// ---- problem constants (B=1, S=2048, H=2048, E=32, K=4, I=768) ----
#define T_TOK 2048
#define HDIM  2048
#define NEXP  32
#define TOPK  4
#define IDIM  768

typedef __attribute__((ext_vector_type(16))) __bf16 v16bf;
typedef __attribute__((ext_vector_type(8)))  float  v8f;

union FragBF {
  v16bf v;
  uint4 q[2];
};

// pack two fp32 -> packed bf16 pair (round-to-nearest-even)
static __device__ __forceinline__ unsigned pk2bf(float lo, float hi) {
  unsigned ul = __float_as_uint(lo);
  unsigned uh = __float_as_uint(hi);
  ul = (ul + 0x7fffu + ((ul >> 16) & 1u)) >> 16;
  uh = (uh + 0x7fffu + ((uh >> 16) & 1u)) & 0xffff0000u;
  return (ul & 0x0000ffffu) | uh;
}

static __device__ __forceinline__ __bf16 f2bf(float f) {
  unsigned u = pk2bf(f, 0.0f);
  unsigned short h = (unsigned short)u;
  __bf16 b;
  __builtin_memcpy(&b, &h, sizeof(b));
  return b;
}

// stage 16 contiguous fp32 -> 16 bf16 in LDS (2x ds_store_b128)
static __device__ __forceinline__ void stage16(__bf16* dst, const float* src) {
  const float4* s = (const float4*)src;
  float4 f0 = s[0], f1 = s[1], f2 = s[2], f3 = s[3];
  uint4 q0, q1;
  q0.x = pk2bf(f0.x, f0.y);
  q0.y = pk2bf(f0.z, f0.w);
  q0.z = pk2bf(f1.x, f1.y);
  q0.w = pk2bf(f1.z, f1.w);
  q1.x = pk2bf(f2.x, f2.y);
  q1.y = pk2bf(f2.z, f2.w);
  q1.z = pk2bf(f3.x, f3.y);
  q1.w = pk2bf(f3.z, f3.w);
  ((uint4*)dst)[0] = q0;
  ((uint4*)dst)[1] = q1;
}

// async DMA: 32 bytes per lane, global -> LDS, tracked by ASYNCcnt.
// INST_OFFSET applies to both LDS and global sides, so one base + offset:16
// moves each lane's two 16B beats.
static __device__ __forceinline__ void async_copy32B(unsigned lds_off,
                                                     const __bf16* src) {
  unsigned long long ga = (unsigned long long)(uintptr_t)src;
  asm volatile("global_load_async_to_lds_b128 %0, %1, off"
               :: "v"(lds_off), "v"(ga) : "memory");
  asm volatile("global_load_async_to_lds_b128 %0, %1, off offset:16"
               :: "v"(lds_off), "v"(ga) : "memory");
}

static __device__ __forceinline__ void wait_async0() {
  asm volatile("s_wait_asynccnt 0x0" ::: "memory");
}

// ---------------------------------------------------------------- zero
__global__ void moe_zero(float* __restrict__ out, int* __restrict__ counts, int n) {
  int i = blockIdx.x * blockDim.x + threadIdx.x;
  if (i < n) out[i] = 0.0f;
  if (i < NEXP) counts[i] = 0;
}

// ---------------------------------------------------------------- router
// one wave32 per token: 32 lanes each own one expert's logit accumulator
__global__ __launch_bounds__(256) void moe_router(
    const float* __restrict__ x, const float* __restrict__ gate_w,
    float* __restrict__ logits_out, int* __restrict__ topk_id,
    float* __restrict__ topk_w) {
  const int lane = threadIdx.x & 31;
  const int wave = threadIdx.x >> 5;
  const int t = blockIdx.x * 8 + wave;
  if (t >= T_TOK) return;

  float acc[NEXP];
#pragma unroll
  for (int e = 0; e < NEXP; ++e) acc[e] = 0.0f;

  const float* xr = x + (size_t)t * HDIM;
  for (int h0 = 0; h0 < HDIM; h0 += 32) {
    float xv = xr[h0 + lane];
#pragma unroll
    for (int e = 0; e < NEXP; ++e)
      acc[e] += xv * gate_w[(size_t)e * HDIM + h0 + lane];
  }

  float logit = 0.0f;
#pragma unroll
  for (int e = 0; e < NEXP; ++e) {
    float s = acc[e];
#pragma unroll
    for (int off = 16; off > 0; off >>= 1) s += __shfl_xor(s, off, 32);
    if (lane == e) logit = s;
  }
  logits_out[(size_t)t * NEXP + lane] = logit;

  // softmax over the 32 lanes
  float m = logit;
#pragma unroll
  for (int off = 16; off > 0; off >>= 1) m = fmaxf(m, __shfl_xor(m, off, 32));
  float p = __expf(logit - m);
  float s = p;
#pragma unroll
  for (int off = 16; off > 0; off >>= 1) s += __shfl_xor(s, off, 32);
  float prob = p / s;

  // top-4 via repeated lane-argmax
  float wsel[TOPK];
  int   isel[TOPK];
  float pc = prob;
#pragma unroll
  for (int k = 0; k < TOPK; ++k) {
    float v = pc;
    int idx = lane;
#pragma unroll
    for (int off = 16; off > 0; off >>= 1) {
      float ov = __shfl_xor(v, off, 32);
      int   oi = __shfl_xor(idx, off, 32);
      if (ov > v || (ov == v && oi < idx)) { v = ov; idx = oi; }
    }
    wsel[k] = v;
    isel[k] = idx;
    if (lane == idx) pc = -1.0f;
  }
  float wsum = wsel[0] + wsel[1] + wsel[2] + wsel[3];
  if (lane < TOPK) {
    topk_id[t * TOPK + lane] = isel[lane];
    topk_w [t * TOPK + lane] = wsel[lane] / wsum;
  }
}

// ---------------------------------------------------------------- histogram / scan / scatter
__global__ void moe_count(const int* __restrict__ topk_id, int* __restrict__ counts) {
  int t = blockIdx.x * blockDim.x + threadIdx.x;
  if (t >= T_TOK) return;
#pragma unroll
  for (int k = 0; k < TOPK; ++k) atomicAdd(&counts[topk_id[t * TOPK + k]], 1);
}

__global__ void moe_scan(const int* __restrict__ counts, int* __restrict__ offsets,
                         int* __restrict__ cursor) {
  if (threadIdx.x == 0) {
    int run = 0;
    for (int e = 0; e < NEXP; ++e) {
      offsets[e] = run;
      cursor[e]  = run;
      run += counts[e];
    }
    offsets[NEXP] = run;
  }
}

__global__ void moe_scatter(const int* __restrict__ topk_id,
                            const float* __restrict__ topk_w,
                            int* __restrict__ cursor, int* __restrict__ slotTok,
                            float* __restrict__ slotW) {
  int t = blockIdx.x * blockDim.x + threadIdx.x;
  if (t >= T_TOK) return;
#pragma unroll
  for (int k = 0; k < TOPK; ++k) {
    int e = topk_id[t * TOPK + k];
    int pos = atomicAdd(&cursor[e], 1);
    slotTok[pos] = t;
    slotW[pos]   = topk_w[t * TOPK + k];
  }
}

// ---------------------------------------------------------------- gate+up GEMM (bf16 WMMA) + SwiGLU
// grid: (IDIM/32, T_TOK/32, NEXP); block: 128 threads = 4 waves, 32x32 tile, K-step 64
__global__ __launch_bounds__(128) void moe_gateup(
    const float* __restrict__ x, const float* __restrict__ gate_proj,
    const float* __restrict__ up_proj, const int* __restrict__ slotTok,
    const int* __restrict__ offsets, __bf16* __restrict__ hact) {
  const int e    = blockIdx.z;
  const int base = offsets[e];
  const int ne   = offsets[e + 1] - base;
  const int m0   = blockIdx.y * 32;
  if (m0 >= ne) return;
  const int n0  = blockIdx.x * 32;
  const int rem = ne - m0;

  // K-major tiles; stride 72 halves = 144 B (16B multiple, bank-friendly)
  __shared__ alignas(16) __bf16 As [32][72];
  __shared__ alignas(16) __bf16 Bgs[32][72];
  __shared__ alignas(16) __bf16 Bus[32][72];
  __shared__ int toks[32];

  const int tid  = threadIdx.x;
  const int lane = tid & 31;
  const int wave = tid >> 5;
  const int wm = (wave >> 1) * 16;  // wave M offset in tile
  const int wn = (wave & 1) * 16;   // wave N offset in tile
  const int g  = lane >> 4;         // lane half
  const int lm = lane & 15;

  if (tid < 32) {
    int rr = (tid < rem) ? tid : (rem - 1);   // clamp: dead rows get a valid token
    toks[tid] = slotTok[base + m0 + rr];
  }
  __syncthreads();

  const int row = tid >> 2;          // 0..31, 4 threads per row
  const int c0  = (tid & 3) * 16;    // 16 contiguous K per thread
  const float* arow = x         + (size_t)toks[row] * HDIM + c0;
  const float* grow = gate_proj + ((size_t)e * IDIM + n0 + row) * HDIM + c0;
  const float* urow = up_proj   + ((size_t)e * IDIM + n0 + row) * HDIM + c0;

  v8f cg = {};
  v8f cu = {};

  for (int k0 = 0; k0 < HDIM; k0 += 64) {
    stage16(&As [row][c0], arow + k0);
    stage16(&Bgs[row][c0], grow + k0);
    stage16(&Bus[row][c0], urow + k0);
    __builtin_prefetch(grow + k0 + 64, 0, 3);
    __builtin_prefetch(urow + k0 + 64, 0, 3);
    __syncthreads();

#pragma unroll
    for (int kk = 0; kk < 2; ++kk) {
      FragBF a, bg, bu;
      // A: lane's elements = K run [16g, 16g+16)
      a.q[0]  = *(const uint4*)&As [wm + lm][kk * 32 + 16 * g];
      a.q[1]  = *(const uint4*)&As [wm + lm][kk * 32 + 16 * g + 8];
      // B ([n][k] in LDS): lane's elements = K run [16g, 16g+16)
      bg.q[0] = *(const uint4*)&Bgs[wn + lm][kk * 32 + 16 * g];
      bg.q[1] = *(const uint4*)&Bgs[wn + lm][kk * 32 + 16 * g + 8];
      bu.q[0] = *(const uint4*)&Bus[wn + lm][kk * 32 + 16 * g];
      bu.q[1] = *(const uint4*)&Bus[wn + lm][kk * 32 + 16 * g + 8];
      cg = __builtin_amdgcn_wmma_f32_16x16x32_bf16(false, a.v, false, bg.v,
                                                   (short)0, cg, false, false);
      cu = __builtin_amdgcn_wmma_f32_16x16x32_bf16(false, a.v, false, bu.v,
                                                   (short)0, cu, false, false);
    }
    __syncthreads();
  }

  // epilogue: hact = silu(gate) * up  -> bf16   (v_rcp_f32, no IEEE div macro)
#pragma unroll
  for (int r = 0; r < 8; ++r) {
    int M = r + 8 * g;               // C/D layout: M = r + 8*(lane>=16)
    int srow = m0 + wm + M;
    if (srow < ne) {
      float gv = cg[r], uv = cu[r];
      float hv = gv * __builtin_amdgcn_rcpf(1.0f + __expf(-gv)) * uv;
      hact[(size_t)(base + srow) * IDIM + (n0 + wn + lm)] = f2bf(hv);
    }
  }
}

// ---------------------------------------------------------------- down GEMM (bf16 WMMA) + weighted combine
// grid: (HDIM/32, T_TOK/32, NEXP); block: 128 threads = 4 waves, 32x32 tile, K-step 64
// A tile (already bf16) is DMA'd with global_load_async_to_lds_b128 (ASYNCcnt).
__global__ __launch_bounds__(128) void moe_down(
    const __bf16* __restrict__ hact, const float* __restrict__ down_proj,
    const int* __restrict__ slotTok, const float* __restrict__ slotW,
    const int* __restrict__ offsets, float* __restrict__ out) {
  const int e    = blockIdx.z;
  const int base = offsets[e];
  const int ne   = offsets[e + 1] - base;
  const int m0   = blockIdx.y * 32;
  if (m0 >= ne) return;
  const int n0  = blockIdx.x * 32;
  const int rem = ne - m0;

  __shared__ alignas(16) __bf16 As[32][72];
  __shared__ alignas(16) __bf16 Bs[32][72];

  const int tid  = threadIdx.x;
  const int lane = tid & 31;
  const int wave = tid >> 5;
  const int wm = (wave >> 1) * 16;
  const int wn = (wave & 1) * 16;
  const int g  = lane >> 4;
  const int lm = lane & 15;

  const int row = tid >> 2;
  const int c0  = (tid & 3) * 16;
  const int ar  = (row < rem) ? row : (rem - 1);  // clamp dead rows
  const __bf16* arow = hact + (size_t)(base + m0 + ar) * IDIM + c0;
  const float*  brow = down_proj + ((size_t)e * HDIM + n0 + row) * IDIM + c0;
  const unsigned aoff = (unsigned)(uintptr_t)&As[row][c0];  // LDS byte offset

  v8f c = {};

  for (int k0 = 0; k0 < IDIM; k0 += 64) {
    async_copy32B(aoff, arow + k0);           // bf16 A tile: DMA, no VGPR traffic
    stage16(&Bs[row][c0], brow + k0);         // fp32 B tile: convert in-flight
    __builtin_prefetch(brow + k0 + 64, 0, 3);
    wait_async0();
    __syncthreads();

#pragma unroll
    for (int kk = 0; kk < 2; ++kk) {
      FragBF a, b;
      a.q[0] = *(const uint4*)&As[wm + lm][kk * 32 + 16 * g];
      a.q[1] = *(const uint4*)&As[wm + lm][kk * 32 + 16 * g + 8];
      b.q[0] = *(const uint4*)&Bs[wn + lm][kk * 32 + 16 * g];
      b.q[1] = *(const uint4*)&Bs[wn + lm][kk * 32 + 16 * g + 8];
      c = __builtin_amdgcn_wmma_f32_16x16x32_bf16(false, a.v, false, b.v,
                                                  (short)0, c, false, false);
    }
    __syncthreads();
  }

#pragma unroll
  for (int r = 0; r < 8; ++r) {
    int M = r + 8 * g;
    int srow = m0 + wm + M;
    if (srow < ne) {
      int   tok = slotTok[base + srow];
      float w   = slotW[base + srow];
      atomicAdd(&out[(size_t)tok * HDIM + (n0 + wn + lm)], c[r] * w);
    }
  }
}

// ---------------------------------------------------------------- host launch
extern "C" void kernel_launch(void* const* d_in, const int* in_sizes, int n_in,
                              void* d_out, int out_size, void* d_ws, size_t ws_size,
                              hipStream_t stream) {
  (void)in_sizes; (void)n_in; (void)out_size; (void)ws_size;
  const float* x         = (const float*)d_in[0];
  const float* gate_w    = (const float*)d_in[1];
  const float* gate_proj = (const float*)d_in[2];
  const float* up_proj   = (const float*)d_in[3];
  const float* down_proj = (const float*)d_in[4];

  float* out    = (float*)d_out;                       // [T*H] fp32
  float* logits = out + (size_t)T_TOK * HDIM;          // [T*E] fp32

  // workspace carve-up
  char* ws = (char*)d_ws;
  size_t off = 0;
  int*   topk_id = (int*)(ws + off);   off += sizeof(int)   * T_TOK * TOPK;
  float* topk_w  = (float*)(ws + off); off += sizeof(float) * T_TOK * TOPK;
  int*   counts  = (int*)(ws + off);   off += sizeof(int)   * NEXP;
  int*   offsets = (int*)(ws + off);   off += sizeof(int)   * (NEXP + 1);
  int*   cursor  = (int*)(ws + off);   off += sizeof(int)   * NEXP;
  off = (off + 63) & ~(size_t)63;
  int*   slotTok = (int*)(ws + off);   off += sizeof(int)   * T_TOK * TOPK;
  float* slotW   = (float*)(ws + off); off += sizeof(float) * T_TOK * TOPK;
  off = (off + 63) & ~(size_t)63;
  __bf16* hact   = (__bf16*)(ws + off);

  const int nz = T_TOK * HDIM;
  moe_zero<<<(nz + 255) / 256, 256, 0, stream>>>(out, counts, nz);
  moe_router<<<T_TOK / 8, 256, 0, stream>>>(x, gate_w, logits, topk_id, topk_w);
  moe_count<<<(T_TOK + 255) / 256, 256, 0, stream>>>(topk_id, counts);
  moe_scan<<<1, 32, 0, stream>>>(counts, offsets, cursor);
  moe_scatter<<<(T_TOK + 255) / 256, 256, 0, stream>>>(topk_id, topk_w, cursor,
                                                       slotTok, slotW);
  moe_gateup<<<dim3(IDIM / 32, T_TOK / 32, NEXP), 128, 0, stream>>>(
      x, gate_proj, up_proj, slotTok, offsets, hact);
  moe_down<<<dim3(HDIM / 32, T_TOK / 32, NEXP), 128, 0, stream>>>(
      hact, down_proj, slotTok, slotW, offsets, out);
}
